// InterpreSegNet_22926535426133
// MI455X (gfx1250) — matrix operand order repaired and checked by reading, not compiled, not run on hardware
//
#include <hip/hip_runtime.h>

// ---------------------------------------------------------------------------
// Types for CDNA5 WMMA / TDM (wave32)
// ---------------------------------------------------------------------------
typedef __attribute__((ext_vector_type(16))) _Float16     v16h;
typedef __attribute__((ext_vector_type(8)))  float        v8f;
typedef __attribute__((ext_vector_type(4)))  unsigned int v4u;
typedef __attribute__((ext_vector_type(8)))  int          v8i;
typedef __attribute__((ext_vector_type(4)))  int          v4i;

#if defined(__gfx1250__) && __has_builtin(__builtin_amdgcn_tensor_load_to_lds) \
    && __has_builtin(__builtin_amdgcn_s_wait_tensorcnt)
#define USE_TDM 1
#else
#define USE_TDM 0
#endif

// Problem constants (from the reference)
#define BATCH    4
#define NCLASSES 2
#define H_IMG    2944
#define W_IMG    1920
#define HC       92
#define WC       60
#define HWC      (HC*WC)          // 5520
#define NCROPS   6
#define NCTOT    (BATCH*NCROPS)   // 24
#define KTOP     110              // int(0.02*92*60)
#define POOLH    85               // 92-8+1
#define POOLW    53               // 60-8+1

#if USE_TDM
// ---------------------------------------------------------------------------
// Issue a TDM 2D tile load: 32 cols x 64 rows of f32 from a row-major
// [OC x K] weight matrix (row stride K) into LDS at ldsOff.
// tensor_dim0/1 are set to the REMAINING extents from the tile origin so the
// TDM's out-of-bounds-reads-return-zero rule implements the K/OC tail masking.
// D# layout per CDNA5 ISA ch.8 (group0 128b, group1 256b; groups 2/3 unused
// for 2D tiles -> zeros).  This toolchain exposes the 6-arg builtin:
//   (uint32x4 g0, int32x8 g1, int32x4 g2, int32x4 g3, int32x8, i32 cpol)
// ---------------------------------------------------------------------------
__device__ __forceinline__ void tdm_load_weights(const float* tile,
                                                 unsigned ldsOff,
                                                 int remK, int remOC, int K)
{
  const unsigned long long ga = (unsigned long long)(size_t)tile;
  v4u g0;
  g0.x = 1u;                                          // count=1, user mode
  g0.y = ldsOff;                                      // lds_addr [63:32]
  g0.z = (unsigned)(ga & 0xFFFFFFFFull);              // global_addr lo
  g0.w = (unsigned)((ga >> 32) & 0x1FFFFFFu)          // global_addr hi (56:32)
         | (2u << 30);                                // type=2 ("image")
  v8i g1;
  g1[0] = (int)(2u << 16);                            // data_size=2 (4 bytes)
  g1[1] = (int)(((unsigned)remK & 0xFFFFu) << 16);    // tensor_dim0[15:0]
  g1[2] = (int)((((unsigned)remK >> 16) & 0xFFFFu)    // tensor_dim0[31:16]
         | (((unsigned)remOC & 0xFFFFu) << 16));      // tensor_dim1[15:0]
  g1[3] = (int)((((unsigned)remOC >> 16) & 0xFFFFu)   // tensor_dim1[31:16]
         | (32u << 16));                              // tile_dim0 = 32
  g1[4] = 64;                                         // tile_dim1 = 64
  g1[5] = K;                                          // tensor_dim0_stride lo
  g1[6] = 0;
  g1[7] = 0;
  const v4i z4 = {0, 0, 0, 0};
  const v8i z8 = {0, 0, 0, 0, 0, 0, 0, 0};
  __builtin_amdgcn_tensor_load_to_lds(g0, g1, z4, z4, z8, 0);
}
#endif

// ---------------------------------------------------------------------------
// Generic 3x3 stride-2 SAME conv + ReLU as implicit GEMM on WMMA f16->f32.
//   M = OC (tile 16), N = 16 output pixels along a row, K = IC*9.
//   Block = 256 threads = 8 waves: 4 OC-tiles x 2 pixel-tiles = 64 OC x 32 px.
//   Weights: TDM DMA into double-buffered LDS, pipelined against the WMMA.
//   B operand: 16 straight-line predicated global loads (im2col gather).
// jax 'SAME', stride 2, k=3, even input: pad_lo=0, pad_hi=1 -> iy = 2*oy+dy.
// ---------------------------------------------------------------------------
__global__ __launch_bounds__(256) void conv3x3_s2_relu_wmma(
    const float* __restrict__ in, const float* __restrict__ wt,
    float* __restrict__ out,
    int N, int IC, int IH, int IW, int OC, int OH, int OW,
    const int* __restrict__ cpos, int srcW, long srcHW)
{
  const int K   = IC * 9;
  const int nch = (K + 31) >> 5;

#if USE_TDM
  __shared__ __align__(16) float wlds[2][64 * 32];
#else
  __shared__ __align__(16) float wlds[1][64 * 32];
#endif

  const int tid      = threadIdx.x;
  const int lane     = tid & 31;
  const int wave     = tid >> 5;
  const int oc_tile  = wave & 3;     // 0..3
  const int px_tile  = wave >> 2;    // 0..1
  const int laneHalf = lane >> 4;    // 0 / 1
  const int lane16   = lane & 15;

  const int b      = blockIdx.y / OH;
  const int oy     = blockIdx.y % OH;
  const int ox     = blockIdx.x * 32 + px_tile * 16 + lane16;
  const int oc_blk = blockIdx.z * 64;

  const int  wrow = (oc_tile * 16 + lane16) * 32;
  const bool oxOK = (ox < OW);
  const int  oy2  = oy * 2;
  const int  ox2  = ox * 2;

  // crop-gather mode (LocalNetwork conv1): read from full image at pos
  long base = 0;
  if (cpos) {
    const int img = b / NCROPS;
    base = (long)img * srcHW + (long)cpos[2 * b] * srcW + (long)cpos[2 * b + 1];
  }

#if USE_TDM
  if (wave == 0) {  // wave-uniform: one TDM issue per workgroup
    tdm_load_weights(wt + (long)oc_blk * K, (unsigned)(size_t)&wlds[0][0],
                     K, OC - oc_blk, K);
    __builtin_amdgcn_s_wait_tensorcnt(0);
  }
  __syncthreads();
#endif

  v8f acc = {};
  for (int c = 0; c < nch; ++c) {
    const int k0 = c << 5;

#if USE_TDM
    if (wave == 0 && c + 1 < nch) {  // prefetch next chunk into other buffer
      tdm_load_weights(wt + (long)oc_blk * K + (k0 + 32),
                       (unsigned)(size_t)&wlds[(c + 1) & 1][0],
                       K - (k0 + 32), OC - oc_blk, K);
    }
    const float* wbuf = wlds[c & 1];
#else
    __syncthreads();
    for (int t = tid; t < 64 * 32; t += 256) {
      const int ocg = oc_blk + (t >> 5);
      const int kg  = k0 + (t & 31);
      wlds[0][t] = (ocg < OC && kg < K) ? wt[(long)ocg * K + kg] : 0.0f;
    }
    __syncthreads();
    const float* wbuf = wlds[0];
#endif

    // ---- A fragment: 4x ds_load_b128 + convert (ISA 7.12.2 16-bit A 16x32:
    //      lanes 0-15 -> K {0..7, 16..23}; lanes 16-31 -> +8) ----
    v16h afrag, bfrag;
    {
      const float4 w0 = *(const float4*)&wbuf[wrow + laneHalf * 8 + 0];
      const float4 w1 = *(const float4*)&wbuf[wrow + laneHalf * 8 + 4];
      const float4 w2 = *(const float4*)&wbuf[wrow + 16 + laneHalf * 8 + 0];
      const float4 w3 = *(const float4*)&wbuf[wrow + 16 + laneHalf * 8 + 4];
      afrag[0]  = (_Float16)w0.x;  afrag[1]  = (_Float16)w0.y;
      afrag[2]  = (_Float16)w0.z;  afrag[3]  = (_Float16)w0.w;
      afrag[4]  = (_Float16)w1.x;  afrag[5]  = (_Float16)w1.y;
      afrag[6]  = (_Float16)w1.z;  afrag[7]  = (_Float16)w1.w;
      afrag[8]  = (_Float16)w2.x;  afrag[9]  = (_Float16)w2.y;
      afrag[10] = (_Float16)w2.z;  afrag[11] = (_Float16)w2.w;
      afrag[12] = (_Float16)w3.x;  afrag[13] = (_Float16)w3.y;
      afrag[14] = (_Float16)w3.z;  afrag[15] = (_Float16)w3.w;
    }

    // ---- B fragment: straight-line predicated im2col gather.
    //      lanes 0-15 hold K {0..7, 16..23}, lanes 16-31 hold K +8; N = lane16. ----
#define CONV_FRAG_B(i)                                                        \
    {                                                                         \
      const int  kg = k0 + laneHalf * 8 + (i) + ((i) >= 8 ? 8 : 0);           \
      const int  ic = kg / 9;                                                 \
      const int  rr = kg - ic * 9;                                            \
      const int  dy = rr / 3;                                                 \
      const int  dx = rr - dy * 3;                                            \
      const int  iy = oy2 + dy;                                               \
      const int  ix = ox2 + dx;                                               \
      const bool ok = (kg < K) && oxOK && (iy < IH) && (ix < IW);             \
      const long addr = ok ? (cpos ? (base + (long)iy * srcW + ix)            \
                                   : (((long)(b * IC + ic) * IH + iy) * IW + ix)) \
                           : 0;                                               \
      const float tv = in[addr];                                              \
      bfrag[i] = (_Float16)(ok ? tv : 0.0f);                                  \
    }
    CONV_FRAG_B(0)  CONV_FRAG_B(1)  CONV_FRAG_B(2)  CONV_FRAG_B(3)
    CONV_FRAG_B(4)  CONV_FRAG_B(5)  CONV_FRAG_B(6)  CONV_FRAG_B(7)
    CONV_FRAG_B(8)  CONV_FRAG_B(9)  CONV_FRAG_B(10) CONV_FRAG_B(11)
    CONV_FRAG_B(12) CONV_FRAG_B(13) CONV_FRAG_B(14) CONV_FRAG_B(15)
#undef CONV_FRAG_B

    acc = __builtin_amdgcn_wmma_f32_16x16x32_f16(
        false, afrag, false, bfrag, (short)0, acc, false, false);

#if USE_TDM
    if (wave == 0) __builtin_amdgcn_s_wait_tensorcnt(0);  // next chunk landed
    __syncthreads();   // data visible; also closes reads of current buffer
#endif
  }

  // --- ReLU + store (C/D layout: VGPR r -> M = r + 8*laneHalf, N = lane16) ---
  if (oxOK) {
#pragma unroll
    for (int r = 0; r < 8; ++r) {
      const int ocg = oc_blk + oc_tile * 16 + r + laneHalf * 8;
      if (ocg < OC) {
        float v = acc[r];
        v = v > 0.0f ? v : 0.0f;
        out[(((long)b * OC + ocg) * OH + oy) * OW + ox] = v;
      }
    }
  }
}

// ---------------------------------------------------------------------------
// 1x1 saliency conv (512 -> 2) + sigmoid + per-pixel class sum.
// Writes logits_map straight into d_out (+24 floats offset).
// ---------------------------------------------------------------------------
__global__ void saliency_kernel(const float* __restrict__ hg,
                                const float* __restrict__ salw,
                                float* __restrict__ out_logits,
                                float* __restrict__ sal,
                                float* __restrict__ ssum)
{
  const int i = blockIdx.x * blockDim.x + threadIdx.x;  // (b, pixel)
  if (i >= BATCH * HWC) return;
  const int b = i / HWC;
  const int p = i - b * HWC;
  const float* h = hg + (long)b * 512 * HWC + p;
  float l0 = 0.0f, l1 = 0.0f;
  for (int c = 0; c < 512; ++c) {
    const float v = h[(long)c * HWC];
    l0 += v * salw[c];
    l1 += v * salw[512 + c];
  }
  out_logits[(long)(b * 2 + 0) * HWC + p] = l0;
  out_logits[(long)(b * 2 + 1) * HWC + p] = l1;
  const float s0 = 1.0f / (1.0f + __expf(-l0));
  const float s1 = 1.0f / (1.0f + __expf(-l1));
  sal[(long)(b * 2 + 0) * HWC + p] = s0;
  sal[(long)(b * 2 + 1) * HWC + p] = s1;
  ssum[(long)b * HWC + p] = s0 + s1;
}

// ---------------------------------------------------------------------------
// Top-110 mean per (b, class): iterative argmax+mask in LDS.
// ---------------------------------------------------------------------------
__global__ __launch_bounds__(256) void topk_kernel(
    const float* __restrict__ sal, float* __restrict__ yglob)
{
  __shared__ float sv[HWC];
  __shared__ float rv[256];
  __shared__ int   ri[256];
  const int bc  = blockIdx.x;        // 0..7 = b*2+c
  const int tid = threadIdx.x;
  for (int i = tid; i < HWC; i += 256) sv[i] = sal[(long)bc * HWC + i];
  __syncthreads();
  float sum = 0.0f;
  for (int it = 0; it < KTOP; ++it) {
    float bv = -1e30f; int bi = 0;
    for (int i = tid; i < HWC; i += 256)
      if (sv[i] > bv) { bv = sv[i]; bi = i; }
    rv[tid] = bv; ri[tid] = bi;
    __syncthreads();
    for (int s = 128; s > 0; s >>= 1) {
      if (tid < s) {
        if (rv[tid + s] > rv[tid] ||
            (rv[tid + s] == rv[tid] && ri[tid + s] < ri[tid])) {
          rv[tid] = rv[tid + s]; ri[tid] = ri[tid + s];
        }
      }
      __syncthreads();
    }
    if (tid == 0) { sum += rv[0]; sv[ri[0]] = -1e30f; }
    __syncthreads();
  }
  if (tid == 0) yglob[bc] = sum * (1.0f / (float)KTOP);
}

// ---------------------------------------------------------------------------
// Greedy ROI retrieval: 6x (argmax of 8x8 box-sum over 85x53, suppress box).
// Emits positions already scaled to full resolution (exact x32).
// ---------------------------------------------------------------------------
__global__ __launch_bounds__(256) void roi_kernel(
    const float* __restrict__ ssum, int* __restrict__ pos)
{
  __shared__ float s[HWC];
  __shared__ float rv[256];
  __shared__ int   ri[256];
  const int b   = blockIdx.x;
  const int tid = threadIdx.x;
  for (int i = tid; i < HWC; i += 256) s[i] = ssum[(long)b * HWC + i];
  __syncthreads();
  for (int crop = 0; crop < NCROPS; ++crop) {
    float bv = -1e30f; int bi = 0;
    for (int wi = tid; wi < POOLH * POOLW; wi += 256) {
      const int wy = wi / POOLW;
      const int wx = wi - wy * POOLW;
      float acc = 0.0f;
      for (int dy = 0; dy < 8; ++dy)
        for (int dx = 0; dx < 8; ++dx)
          acc += s[(wy + dy) * WC + wx + dx];
      if (acc > bv) { bv = acc; bi = wi; }  // strict > keeps first index
    }
    rv[tid] = bv; ri[tid] = bi;
    __syncthreads();
    for (int st = 128; st > 0; st >>= 1) {
      if (tid < st) {
        if (rv[tid + st] > rv[tid] ||
            (rv[tid + st] == rv[tid] && ri[tid + st] < ri[tid])) {
          rv[tid] = rv[tid + st]; ri[tid] = ri[tid + st];
        }
      }
      __syncthreads();
    }
    const int wy = ri[0] / POOLW;
    const int wx = ri[0] - wy * POOLW;
    if (tid == 0) {
      pos[(b * NCROPS + crop) * 2 + 0] = wy * 32;  // 2944/92 == 32 exactly
      pos[(b * NCROPS + crop) * 2 + 1] = wx * 32;  // 1920/60 == 32 exactly
    }
    if (tid < 64) s[(wy + (tid >> 3)) * WC + wx + (tid & 7)] = -2.0f;
    __syncthreads();
  }
}

// ---------------------------------------------------------------------------
// Global average pool 16x16 -> h_crops[24,512]
// ---------------------------------------------------------------------------
__global__ void gap_kernel(const float* __restrict__ c4,
                           float* __restrict__ hc)
{
  const int i = blockIdx.x * blockDim.x + threadIdx.x;  // crop*512+ch
  if (i >= NCTOT * 512) return;
  const float* p = c4 + (long)i * 256;
  float s = 0.0f;
  for (int j = 0; j < 256; ++j) s += p[j];
  hc[i] = s * (1.0f / 256.0f);
}

// ---------------------------------------------------------------------------
// Global max pool over 92x60 -> global_vec[4,512]
// ---------------------------------------------------------------------------
__global__ void gmax_kernel(const float* __restrict__ hg,
                            float* __restrict__ gv)
{
  const int i = blockIdx.x * blockDim.x + threadIdx.x;  // b*512+ch
  if (i >= BATCH * 512) return;
  const float* p = hg + (long)i * HWC;
  float m = p[0];
  for (int j = 1; j < HWC; ++j) m = fmaxf(m, p[j]);
  gv[i] = m;
}

// ---------------------------------------------------------------------------
// Gated-attention MIL + classifier + fusion heads (single block, deterministic)
// ---------------------------------------------------------------------------
__global__ __launch_bounds__(256) void attn_fusion_kernel(
    const float* __restrict__ hc, const float* __restrict__ attV,
    const float* __restrict__ attU, const float* __restrict__ attw,
    const float* __restrict__ clsw, const float* __restrict__ clsb,
    const float* __restrict__ fusw, const float* __restrict__ fusb,
    const float* __restrict__ gv, float* __restrict__ out)
{
  __shared__ float gates[NCTOT * 128];
  __shared__ float score[NCTOT];
  __shared__ float attn[NCTOT];
  __shared__ float zsh[BATCH * 512];
  const int tid = threadIdx.x;

  // gate(pk, a) = tanh(h.V) * sigmoid(h.U) * w[a]
  for (int t = tid; t < NCTOT * 128; t += 256) {
    const int pk = t >> 7;
    const int a  = t & 127;
    const float* h = hc + pk * 512;
    float gvv = 0.0f, gu = 0.0f;
    for (int d = 0; d < 512; ++d) {
      const float hv = h[d];
      gvv += hv * attV[d * 128 + a];
      gu  += hv * attU[d * 128 + a];
    }
    gates[t] = tanhf(gvv) * (1.0f / (1.0f + __expf(-gu))) * attw[a];
  }
  __syncthreads();
  if (tid < NCTOT) {
    float s = 0.0f;
    for (int a = 0; a < 128; ++a) s += gates[tid * 128 + a];
    score[tid] = s;
  }
  __syncthreads();
  if (tid < BATCH) {  // softmax over the 6 crops
    float mx = score[tid * NCROPS];
    for (int k = 1; k < NCROPS; ++k) mx = fmaxf(mx, score[tid * NCROPS + k]);
    float den = 0.0f, e[NCROPS];
    for (int k = 0; k < NCROPS; ++k) {
      e[k] = __expf(score[tid * NCROPS + k] - mx);
      den += e[k];
    }
    for (int k = 0; k < NCROPS; ++k) attn[tid * NCROPS + k] = e[k] / den;
  }
  __syncthreads();
  for (int t = tid; t < BATCH * 512; t += 256) {
    const int b = t >> 9;
    const int d = t & 511;
    float z = 0.0f;
    for (int k = 0; k < NCROPS; ++k)
      z += attn[b * NCROPS + k] * hc[(b * NCROPS + k) * 512 + d];
    zsh[t] = z;
  }
  __syncthreads();
  if (tid < 8) {
    const int b = tid >> 1;
    const int c = tid & 1;
    float yl = clsb[c], yf = fusb[c];
    for (int d = 0; d < 512; ++d) {
      yl += zsh[b * 512 + d] * clsw[d * 2 + c];
      yf += gv[b * 512 + d] * fusw[d * 2 + c]
          + zsh[b * 512 + d] * fusw[(512 + d) * 2 + c];
    }
    out[8  + b * 2 + c] = 1.0f / (1.0f + __expf(-yl));  // y_local
    out[16 + b * 2 + c] = 1.0f / (1.0f + __expf(-yf));  // y_fusion
  }
}

// ---------------------------------------------------------------------------
// Workspace layout (float elements). bufA / bufB are ping-ponged activations.
//   bufA: max(a1=4*16*1472*960, a3, c1) = 90,439,680
//   bufB: max(a2=4*32*736*480,  a4, c2) = 45,219,840
// ---------------------------------------------------------------------------
static const size_t OFF_BUFA = 0;
static const size_t OFF_BUFB = OFF_BUFA + 90439680ull;
static const size_t OFF_HG   = OFF_BUFB + 45219840ull;
static const size_t OFF_SAL  = OFF_HG   + 11304960ull;   // sigmoid saliency
static const size_t OFF_SSUM = OFF_SAL  + 44160ull;      // class-summed saliency
static const size_t OFF_POS  = OFF_SSUM + 22080ull;      // 48 ints
static const size_t OFF_HCR  = OFF_POS  + 64ull;         // h_crops [24,512]
static const size_t OFF_GV   = OFF_HCR  + 12288ull;      // global_vec [4,512]

extern "C" void kernel_launch(void* const* d_in, const int* in_sizes, int n_in,
                              void* d_out, int out_size, void* d_ws, size_t ws_size,
                              hipStream_t stream)
{
  const float* x    = (const float*)d_in[0];
  const float* gw1  = (const float*)d_in[1];
  const float* gw2  = (const float*)d_in[2];
  const float* gw3  = (const float*)d_in[3];
  const float* gw4  = (const float*)d_in[4];
  const float* gw5  = (const float*)d_in[5];
  const float* salw = (const float*)d_in[6];
  const float* lw1  = (const float*)d_in[7];
  const float* lw2  = (const float*)d_in[8];
  const float* lw3  = (const float*)d_in[9];
  const float* lw4  = (const float*)d_in[10];
  const float* attV = (const float*)d_in[11];
  const float* attU = (const float*)d_in[12];
  const float* attw = (const float*)d_in[13];
  const float* clsw = (const float*)d_in[14];
  const float* clsb = (const float*)d_in[15];
  const float* fusw = (const float*)d_in[16];
  const float* fusb = (const float*)d_in[17];

  float* ws   = (float*)d_ws;
  float* bufA = ws + OFF_BUFA;
  float* bufB = ws + OFF_BUFB;
  float* hg   = ws + OFF_HG;
  float* sal  = ws + OFF_SAL;
  float* ssum = ws + OFF_SSUM;
  int*   pos  = (int*)(ws + OFF_POS);
  float* hcr  = ws + OFF_HCR;
  float* gvec = ws + OFF_GV;
  float* out  = (float*)d_out;

  const dim3 blk(256);
  #define CONV_GRID(OW_, OH_, N_, OC_) \
      dim3(((OW_) + 31) / 32, (N_) * (OH_), ((OC_) + 63) / 64)

  // ---- GlobalNetwork (implicit-GEMM WMMA convs, ping-pong bufA/bufB) ----
  conv3x3_s2_relu_wmma<<<CONV_GRID(960, 1472, 4, 16), blk, 0, stream>>>(
      x, gw1, bufA, 4, 1, 2944, 1920, 16, 1472, 960, nullptr, 0, 0);
  conv3x3_s2_relu_wmma<<<CONV_GRID(480, 736, 4, 32), blk, 0, stream>>>(
      bufA, gw2, bufB, 4, 16, 1472, 960, 32, 736, 480, nullptr, 0, 0);
  conv3x3_s2_relu_wmma<<<CONV_GRID(240, 368, 4, 64), blk, 0, stream>>>(
      bufB, gw3, bufA, 4, 32, 736, 480, 64, 368, 240, nullptr, 0, 0);
  conv3x3_s2_relu_wmma<<<CONV_GRID(120, 184, 4, 128), blk, 0, stream>>>(
      bufA, gw4, bufB, 4, 64, 368, 240, 128, 184, 120, nullptr, 0, 0);
  conv3x3_s2_relu_wmma<<<CONV_GRID(60, 92, 4, 512), blk, 0, stream>>>(
      bufB, gw5, hg, 4, 128, 184, 120, 512, 92, 60, nullptr, 0, 0);

  // ---- Saliency head + top-t% aggregation + ROI retrieval ----
  saliency_kernel<<<(BATCH * HWC + 255) / 256, blk, 0, stream>>>(
      hg, salw, out + 24, sal, ssum);
  topk_kernel<<<BATCH * NCLASSES, blk, 0, stream>>>(sal, out);
  roi_kernel<<<BATCH, blk, 0, stream>>>(ssum, pos);

  // ---- LocalNetwork (conv1 fused with crop gather) ----
  conv3x3_s2_relu_wmma<<<CONV_GRID(128, 128, NCTOT, 64), blk, 0, stream>>>(
      x, lw1, bufA, NCTOT, 1, 256, 256, 64, 128, 128,
      pos, W_IMG, (long)H_IMG * W_IMG);
  conv3x3_s2_relu_wmma<<<CONV_GRID(64, 64, NCTOT, 128), blk, 0, stream>>>(
      bufA, lw2, bufB, NCTOT, 64, 128, 128, 128, 64, 64, nullptr, 0, 0);
  conv3x3_s2_relu_wmma<<<CONV_GRID(32, 32, NCTOT, 256), blk, 0, stream>>>(
      bufB, lw3, bufA, NCTOT, 128, 64, 64, 256, 32, 32, nullptr, 0, 0);
  conv3x3_s2_relu_wmma<<<CONV_GRID(16, 16, NCTOT, 512), blk, 0, stream>>>(
      bufA, lw4, bufB, NCTOT, 256, 32, 32, 512, 16, 16, nullptr, 0, 0);

  // ---- pooling + attention MIL + fusion ----
  gap_kernel<<<(NCTOT * 512 + 255) / 256, blk, 0, stream>>>(bufB, hcr);
  gmax_kernel<<<(BATCH * 512 + 255) / 256, blk, 0, stream>>>(hg, gvec);
  attn_fusion_kernel<<<1, blk, 0, stream>>>(
      hcr, attV, attU, attw, clsw, clsb, fusw, fusb, gvec, out);
}
